// GaussianBW_58677843198012
// MI455X (gfx1250) — matrix operand order repaired and checked by reading, not compiled, not run on hardware
//
#include <hip/hip_runtime.h>
#include <hip/hip_bf16.h>
#include <math.h>

typedef __attribute__((ext_vector_type(16))) _Float16 v16h;
typedef __attribute__((ext_vector_type(8)))  float    v8f;

#define GN 16384
#define GH 512
#define GW 512
#define GC 3
#define STAGE 256   // gaussians staged into LDS per pass
#define WAVES 8     // waves per block (256 threads, wave32)

// ---------------------------------------------------------------------------
// CDNA5 async global->LDS copy helpers (ASYNCcnt-tracked, see 08_async_tensor)
// VDST operand = 32-bit LDS byte address, VADDR = 64-bit global address.
// ---------------------------------------------------------------------------
__device__ __forceinline__ void async_copy_b128(void* lds_dst, const void* gsrc) {
  unsigned ldsoff = (unsigned)(size_t)lds_dst;
  asm volatile("global_load_async_to_lds_b128 %0, %1, off"
               :: "v"(ldsoff), "v"(gsrc) : "memory");
}
__device__ __forceinline__ void async_copy_b32(void* lds_dst, const void* gsrc) {
  unsigned ldsoff = (unsigned)(size_t)lds_dst;
  asm volatile("global_load_async_to_lds_b32 %0, %1, off"
               :: "v"(ldsoff), "v"(gsrc) : "memory");
}
__device__ __forceinline__ void wait_async0() {
  asm volatile("s_wait_asynccnt 0" ::: "memory");
}

// ---------------------------------------------------------------------------
// Prep: fold tanh/sigmoid/sincos + covariance inversion into per-gaussian
// constants. G[2n] = (cx, cy, cA, cB), G[2n+1] = (cC, op, 0, 0),
// CH[c][n] = f16 color (channel-major for B-fragment staging).
// ---------------------------------------------------------------------------
__global__ __launch_bounds__(256) void GaussianBW_prep(
    const float* __restrict__ xy, const float* __restrict__ scaling,
    const float* __restrict__ rotation, const float* __restrict__ values,
    const float* __restrict__ opacity,
    float4* __restrict__ G, _Float16* __restrict__ CH)
{
  int i = blockIdx.x * blockDim.x + threadIdx.x;
  if (i >= GN) return;
  float mx = tanhf(xy[2*i+0]);
  float my = tanhf(xy[2*i+1]);
  float sx = fabsf(scaling[2*i+0]) + 0.3f;
  float sy = fabsf(scaling[2*i+1]) + 0.3f;
  float th = 6.28318530717958647692f / (1.0f + __expf(-rotation[i]));
  float ct = __cosf(th);
  float st = __sinf(th);
  float a = ct*sx, b = -st*sy, c = st*sx, d = ct*sy;
  float S00 = a*a + b*b;
  float S01 = a*c + b*d;
  float S11 = c*c + d*d;
  float inv = 1.0f / (S00*S11 - S01*S01);
  float cA = S11*inv, cB = -S01*inv, cC = S00*inv;
  float cx = 0.5f*(float)GW*(mx + 1.0f) - 0.5f;
  float cy = 0.5f*(float)GH*(my + 1.0f) - 0.5f;
  G[2*i+0] = make_float4(cx, cy, cA, cB);
  G[2*i+1] = make_float4(cC, opacity[i], 0.0f, 0.0f);
  CH[0*GN + i] = (_Float16)values[3*i+0];
  CH[1*GN + i] = (_Float16)values[3*i+1];
  CH[2*GN + i] = (_Float16)values[3*i+2];
}

// ---------------------------------------------------------------------------
// Render: one wave per 16-pixel strip. Alphas computed in f32 VALU, packed
// into a f16 WMMA A-fragment (16 px x 32 gaussians); B-fragment = colors
// (32 x 16, cols 0..2 live); f32 16x16 accumulator carries all 512 K-chunks.
// Gaussian params/colors are double-buffered in LDS via async global->LDS
// copies so the TDM fill of pass p+1 overlaps the VALU/WMMA work of pass p.
// ---------------------------------------------------------------------------
__global__ __launch_bounds__(256) void GaussianBW_render(
    const float4* __restrict__ G, const _Float16* __restrict__ CH,
    const float* __restrict__ background, float* __restrict__ out)
{
  __shared__ float4    sP[2][STAGE * 2];
  __shared__ _Float16  sCol[2][3][STAGE];

  const int  tid  = threadIdx.x;
  const int  lane = tid & 31;
  const bool hi   = lane >= 16;
  const int  n    = lane & 15;          // A: pixel M; B/D: column N

  const int   sid  = blockIdx.x * WAVES + (tid >> 5);   // strip id
  const int   py_i = sid >> 5;                          // image row
  const int   px0  = (sid & 31) << 4;                   // strip x base
  const float px   = (float)(px0 + n);
  const float py   = (float)py_i;

  // async-stage one 256-gaussian block into LDS buffer `buf`
  auto stage = [&](int buf, int base) {
    async_copy_b128(&sP[buf][2*tid+0], &G[2*(base + tid) + 0]);
    async_copy_b128(&sP[buf][2*tid+1], &G[2*(base + tid) + 1]);
    for (int i = tid; i < 3*(STAGE/2); i += 256) {
      int c = i / (STAGE/2);
      int p = i - c*(STAGE/2);
      async_copy_b32(((unsigned int*)&sCol[buf][c][0]) + p,
                     ((const unsigned int*)(CH + c*GN + base)) + p);
    }
  };

  v8f acc = {};

  const int NPASS = GN / STAGE;
  stage(0, 0);
  for (int pass = 0; pass < NPASS; ++pass) {
    const int buf = pass & 1;
    wait_async0();        // my async fills of `buf` have landed in LDS
    __syncthreads();      // everyone's fills landed; prev buffer fully read
    if (pass + 1 < NPASS) stage(buf ^ 1, (pass + 1) * STAGE);

    for (int ch = 0; ch < STAGE; ch += 32) {
      // ---- A fragment: 16x32 f16, ISA layout for V_WMMA_*_16X16X32 ----
      v16h afrag;
      #pragma unroll
      for (int v = 0; v < 8; ++v) {
        int k = ch + ((v < 4) ? 2*v : 2*v + 8) + (hi ? 8 : 0);
        float4 p0 = sP[buf][2*k+0];
        float4 p1 = sP[buf][2*k+1];
        float4 q0 = sP[buf][2*k+2];
        float4 q1 = sP[buf][2*k+3];

        float dx  = p0.x - px, dy  = p0.y - py;
        float sg  = 0.5f*(p0.z*dx*dx + p1.x*dy*dy) + p0.w*dx*dy;
        float al  = p1.y * __expf(-sg);
        al = fminf(al, 0.999f);
        al = (sg >= 0.0f && al >= (1.0f/255.0f)) ? al : 0.0f;

        float dx2 = q0.x - px, dy2 = q0.y - py;
        float sg2 = 0.5f*(q0.z*dx2*dx2 + q1.x*dy2*dy2) + q0.w*dx2*dy2;
        float al2 = q1.y * __expf(-sg2);
        al2 = fminf(al2, 0.999f);
        al2 = (sg2 >= 0.0f && al2 >= (1.0f/255.0f)) ? al2 : 0.0f;

        afrag[2*v+0] = (_Float16)al;
        afrag[2*v+1] = (_Float16)al2;
      }

      // ---- B fragment: 32x16 f16, columns 0..2 = RGB, rest zero ----
      v16h bfrag;
      {
        const int      nc   = (n < 3) ? n : 0;
        const _Float16 zero = (_Float16)0.0f;
        #pragma unroll
        for (int j = 0; j < 8; ++j) {
          int kb = ch + ((j < 4) ? 2*j : 2*j + 8) + (hi ? 8 : 0);
          _Float16 c0 = sCol[buf][nc][kb+0];
          _Float16 c1 = sCol[buf][nc][kb+1];
          bfrag[2*j+0] = (n < 3) ? c0 : zero;
          bfrag[2*j+1] = (n < 3) ? c1 : zero;
        }
      }

      acc = __builtin_amdgcn_wmma_f32_16x16x32_f16(
          false, afrag, false, bfrag, (short)0, acc, false, false);
    }
  }

  // D layout: lane holds column N = n, rows M = v + (hi?8:0)
  if (n < 3) {
    float bg = background[n];
    int   mb = hi ? 8 : 0;
    #pragma unroll
    for (int v = 0; v < 8; ++v) {
      int x = px0 + mb + v;
      out[n*(GH*GW) + py_i*GW + x] = acc[v] + bg;
    }
  }
}

extern "C" void kernel_launch(void* const* d_in, const int* in_sizes, int n_in,
                              void* d_out, int out_size, void* d_ws, size_t ws_size,
                              hipStream_t stream) {
  const float* xy       = (const float*)d_in[0];
  const float* scaling  = (const float*)d_in[1];
  const float* rotation = (const float*)d_in[2];
  const float* values   = (const float*)d_in[3];
  const float* opacity  = (const float*)d_in[4];
  const float* backgnd  = (const float*)d_in[5];
  float* out = (float*)d_out;

  float4*   G  = (float4*)d_ws;                                         // GN*2 float4
  _Float16* CH = (_Float16*)((char*)d_ws + (size_t)GN*2*sizeof(float4)); // 3*GN f16

  GaussianBW_prep<<<GN/256, 256, 0, stream>>>(xy, scaling, rotation, values,
                                              opacity, G, CH);

  const int strips = GH * (GW/16);          // 16384
  GaussianBW_render<<<strips/WAVES, 256, 0, stream>>>(G, CH, backgnd, out);
}
